// Module_89283780149523
// MI455X (gfx1250) — matrix operand order repaired and checked by reading, not compiled, not run on hardware
//
#include <hip/hip_runtime.h>
#include <stdint.h>

typedef __attribute__((ext_vector_type(16))) _Float16 v16h;
typedef __attribute__((ext_vector_type(8)))  float    v8f;

#define B_    64
#define T_    400
#define N_    1024
#define NIN_  128
#define NOUT_ 32
#define D_    3
#define L_    5
#define KTOT_ (NIN_ + D_*N_)     /* 3200 */
#define NC_   (KTOT_ / 32)       /* 100 K-chunks */
#define BN_   (B_*N_)            /* 65536 */
#define LDT   40                 /* padded LDS row stride in halves (80B rows, 16B aligned) */

// ---- workspace layout (bytes) ----
static constexpr size_t OFF_BMAT = 0;
static constexpr size_t SZ_BMAT  = (size_t)KTOT_ * N_ * sizeof(_Float16);   // 6.5 MB f16, [o][k] (transposed)
static constexpr size_t OFF_MEM  = OFF_BMAT + SZ_BMAT;
static constexpr size_t SZ_MEM   = (size_t)BN_ * 4;
static constexpr size_t OFF_WP   = OFF_MEM + SZ_MEM;
static constexpr size_t OFF_BUFO = OFF_WP + SZ_MEM;
static constexpr size_t SZ_BUF   = (size_t)L_ * BN_ * 4;
static constexpr size_t OFF_BUFW = OFF_BUFO + SZ_BUF;
static constexpr size_t OFF_REC  = OFF_BUFW + SZ_BUF;
static constexpr size_t SZ_REC   = (size_t)T_ * BN_;                        // u8 spikes
static constexpr size_t OFF_H2   = OFF_REC + SZ_REC;
static constexpr size_t STATE_FLOATS = (2*SZ_MEM + 2*SZ_BUF) / 4;           // mem, w_p, buf_out, buf_wp

// ---- zero the recurrent state (mem, w_p, ring buffers) ----
__global__ void zero_state(float* __restrict__ ptr, int n) {
  int g = blockIdx.x * blockDim.x + threadIdx.x;
  if (g < n) ptr[g] = 0.0f;
}

// ---- build fused f16 B matrix, TRANSPOSED [o][k]: k<128 = w_in, k>=128 = dmap*sign*|w| per delay ----
__global__ void prep_B(const float* __restrict__ w, const float* __restrict__ w_in,
                       const float* __restrict__ w_signs, const float* __restrict__ dmap,
                       _Float16* __restrict__ BmatT) {
  int gid = blockIdx.x * blockDim.x + threadIdx.x;
  if (gid >= N_ * KTOT_) return;
  int o = gid / KTOT_;
  int k = gid - o * KTOT_;
  float v;
  if (k < NIN_) {
    v = w_in[k * N_ + o];
  } else {
    int ks = k - NIN_;
    int d = ks >> 10;
    int e = ks & (N_ - 1);
    v = dmap[((size_t)d * N_ + e) * N_ + o] * w_signs[e] * fabsf(w[(size_t)e * N_ + o]);
  }
  BmatT[gid] = (_Float16)v;   // [o][k], row length 3200 halves (6400 B, 16B-aligned rows)
}

// ---- one recurrent time step, fully fused, double-buffered WMMA pipeline (256 thr / 8 waves) ----
__global__ __launch_bounds__(256) void step_k(
    const float* __restrict__ inputs, const float* __restrict__ p,
    const int* __restrict__ delays, const _Float16* __restrict__ BmatT,
    float* __restrict__ mem, float* __restrict__ wp,
    float* __restrict__ bufo, float* __restrict__ bufw,
    unsigned char* __restrict__ rec, int t)
{
  __shared__ _Float16 ldsA[2][64 * LDT];
  __shared__ _Float16 ldsB[2][64 * LDT];
  const int tid = threadIdx.x;
  const int n0  = blockIdx.x * 64;          // 16 blocks tile N
  const int r0  = t % L_;

  int rid[D_];
#pragma unroll
  for (int d = 0; d < D_; ++d) {
    int dl = delays[d];
    rid[d] = ((t - dl) % L_ + L_) % L_;     // never equals r0 (min delay 1, L=5)
  }

  // Phase 1: spike, record into ring buffers, plasticity update, u8 spike record.
  for (int i = tid; i < B_ * 64; i += 256) {
    int b = i >> 6;
    int n = n0 + (i & 63);
    int idx = b * N_ + n;
    float m  = mem[idx];
    float o  = (m > 1.0f) ? 1.0f : 0.0f;    // spike(mem - 1)
    float w  = wp[idx];
    bufo[r0 * BN_ + idx] = o;
    bufw[r0 * BN_ + idx] = w;
    float pn   = p[n];
    float depr = (pn < 0.0f) ? 1.0f : 0.0f;
    wp[idx] = w * 0.99f + o * pn * (1.0f + depr * w);
    rec[(size_t)t * BN_ + idx] = (unsigned char)o;
  }

  // Phase 2: GEMM  C[b,o] = sum_k A[b,k] * B[k, n0+o],  K = 3200, f16 WMMA / f32 acc
  // 8 waves: wave owns a 16(M) x 32(N) strip -> 2 accumulators, low register pressure (no spills)
  const int wid    = tid >> 5;
  const int lane   = tid & 31;
  const int hh     = lane >> 4;    // lane half (CDNA5 16-bit WMMA layout)
  const int ln     = lane & 15;
  const int mstrip = wid >> 1;     // 0..3 -> M rows 16*mstrip
  const int nhalf  = wid & 1;      // 0..1 -> N cols 32*nhalf

  v8f acc[2];
  {
    v8f z = {0.f, 0.f, 0.f, 0.f, 0.f, 0.f, 0.f, 0.f};
    acc[0] = z; acc[1] = z;
  }

  float2 fa[4];                    // staged A pairs (input value, or buf_out pair)
  float2 fw[4] = {};               // staged buf_wp pairs (spike region only)
  uint4  rb;                       // staged B 8-half vector

  // fetch chunk k0 into registers (vmem only; no waits forced here)
  auto fetch = [&](int k0) {
    const bool inreg = (k0 < NIN_);
#pragma unroll
    for (int j = 0; j < 4; ++j) {
      int pid = tid + j * 256;           // 1024 half-pairs: A tile is 64(b) x 32(k)
      int b   = pid >> 4;
      int kk  = (pid & 15) << 1;
      int kg  = k0 + kk;
      if (inreg) {
        fa[j] = *reinterpret_cast<const float2*>(&inputs[((size_t)b * T_ + t) * NIN_ + kg]);
      } else {
        int ks = kg - NIN_;
        int d = ks >> 10;
        int e = ks & (N_ - 1);
        int off = rid[d] * BN_ + b * N_ + e;
        fa[j] = *reinterpret_cast<const float2*>(&bufo[off]);
        fw[j] = *reinterpret_cast<const float2*>(&bufw[off]);
      }
    }
    {
      int ol = tid >> 2, q = tid & 3;    // 256 uint4: B tile is 64(o) x 32(k), 4 uint4 per row
      rb = *reinterpret_cast<const uint4*>(&BmatT[(size_t)(n0 + ol) * KTOT_ + k0 + 8 * q]);
    }
  };

  // convert + store staged registers into LDS buffer bi
  auto flush = [&](int k0, int bi) {
    const bool inreg = (k0 < NIN_);
#pragma unroll
    for (int j = 0; j < 4; ++j) {
      int pid = tid + j * 256;
      int b   = pid >> 4;
      int kk  = (pid & 15) << 1;
      float v0, v1;
      if (inreg) { v0 = fa[j].x;                      v1 = fa[j].y; }
      else       { v0 = fa[j].x * (1.0f + fw[j].x);   v1 = fa[j].y * (1.0f + fw[j].y); }
      union { _Float16 h[2]; unsigned u; } pk;
      pk.h[0] = (_Float16)v0;
      pk.h[1] = (_Float16)v1;
      *reinterpret_cast<unsigned*>(&ldsA[bi][b * LDT + kk]) = pk.u;
    }
    {
      int ol = tid >> 2, q = tid & 3;
      *reinterpret_cast<uint4*>(&ldsB[bi][ol * LDT + 8 * q]) = rb;
    }
  };

  // prologue: stage chunk 0
  fetch(0);
  flush(0, 0);
  __syncthreads();

  union Frag { uint4 u[2]; v16h v; };

  for (int kc = 0; kc < NC_; ++kc) {
    const int  cur       = kc & 1;
    const bool have_next = (kc + 1 < NC_);
    if (have_next) fetch((kc + 1) * 32);                 // vmem issued early, overlapped with WMMA
    if (kc + 2 < NC_ && tid < 64)                        // keep chunk+2 weight rows warm in L2/L0
      __builtin_prefetch(&BmatT[(size_t)(n0 + tid) * KTOT_ + (kc + 2) * 32], 0, 1);

    // fragment loads from current buffer first, so WMMA's ds wait excludes next-tile stores
    Frag af;
    {
      // A k order chosen here (not the ISA table order): lane half hh -> k = 16hh..16hh+15 (matches B)
      const _Float16* ap = &ldsA[cur][(mstrip * 16 + ln) * LDT];
      af.u[0] = *reinterpret_cast<const uint4*>(ap + 16 * hh);
      af.u[1] = *reinterpret_cast<const uint4*>(ap + 16 * hh + 8);
    }
    Frag bf[2];
#pragma unroll
    for (int s = 0; s < 2; ++s) {
      // B k order chosen here (same as A): lane half hh -> k = 16hh..16hh+15, col = ln
      const _Float16* bp = &ldsB[cur][(nhalf * 32 + s * 16 + ln) * LDT + 16 * hh];
      bf[s].u[0] = *reinterpret_cast<const uint4*>(bp);
      bf[s].u[1] = *reinterpret_cast<const uint4*>(bp + 8);
    }
#pragma unroll
    for (int s = 0; s < 2; ++s) {
      acc[s] = __builtin_amdgcn_wmma_f32_16x16x32_f16(
          false, af.v, false, bf[s].v, (short)0, acc[s], false, false);
    }

    if (have_next) flush((kc + 1) * 32, cur ^ 1);        // waits vmem here, after WMMAs issued
    __syncthreads();
  }

  // Phase 3: mem update  mem = 0.95*mem + (inp_t + syn) - out
#pragma unroll
  for (int s = 0; s < 2; ++s) {
#pragma unroll
    for (int r = 0; r < 8; ++r) {
      int b = mstrip * 16 + r + 8 * hh;     // C/D layout: VGPR r -> M = r + 8*half
      int o = n0 + nhalf * 32 + s * 16 + ln;
      int idx = b * N_ + o;
      float ov = bufo[r0 * BN_ + idx];
      mem[idx] = 0.95f * mem[idx] + acc[s][r] - ov;
    }
  }
}

// ---- readout stage 1: h2[t,b,j] = sum_n rec[t,b,n] * w_out[n,j]  (one wave per (t,b)) ----
__global__ __launch_bounds__(128) void readout1(const unsigned char* __restrict__ rec,
                                                const float* __restrict__ w_out,
                                                float* __restrict__ h2) {
  int wid  = threadIdx.x >> 5;
  int lane = threadIdx.x & 31;                 // lane == output channel j (NOUT=32)
  int pair = blockIdx.x * 4 + wid;             // pair = t*64 + b, 0..25599
  const unsigned char* rp = rec + (size_t)pair * N_;
  float h = 0.0f;
  for (int n = 0; n < N_; ++n) {
    float sp = (float)rp[n];
    h = fmaf(sp, w_out[n * NOUT_ + lane], h);
  }
  h2[(size_t)pair * NOUT_ + lane] = h;
}

// ---- readout stage 2: leaky scan r = 0.95 r + h2, emit (B,T,NOUT) ----
__global__ void readout2(const float* __restrict__ h2, float* __restrict__ out) {
  int gid = blockIdx.x * blockDim.x + threadIdx.x;
  if (gid >= B_ * NOUT_) return;
  int b = gid >> 5, j = gid & 31;
  float r = 0.0f;
  for (int t = 0; t < T_; ++t) {
    r = 0.95f * r + h2[((size_t)t * B_ + b) * NOUT_ + j];
    out[((size_t)b * T_ + t) * NOUT_ + j] = r;
  }
}

extern "C" void kernel_launch(void* const* d_in, const int* in_sizes, int n_in,
                              void* d_out, int out_size, void* d_ws, size_t ws_size,
                              hipStream_t stream) {
  (void)in_sizes; (void)n_in; (void)out_size; (void)ws_size;
  const float* inputs  = (const float*)d_in[0];
  const float* w       = (const float*)d_in[1];
  const float* w_in    = (const float*)d_in[2];
  const float* w_out   = (const float*)d_in[3];
  const float* w_signs = (const float*)d_in[4];
  const float* dmap    = (const float*)d_in[5];
  const float* p       = (const float*)d_in[6];
  const int*   delays  = (const int*)d_in[7];
  float* out = (float*)d_out;

  char* ws = (char*)d_ws;
  _Float16*      BmatT = (_Float16*)(ws + OFF_BMAT);
  float*         mem   = (float*)(ws + OFF_MEM);
  float*         wp    = (float*)(ws + OFF_WP);
  float*         bufo  = (float*)(ws + OFF_BUFO);
  float*         bufw  = (float*)(ws + OFF_BUFW);
  unsigned char* rec   = (unsigned char*)(ws + OFF_REC);
  float*         h2    = (float*)(ws + OFF_H2);

  // zero recurrent state every call (deterministic; ws is not re-poisoned between replays)
  {
    int n = (int)STATE_FLOATS;
    zero_state<<<(n + 255) / 256, 256, 0, stream>>>(mem, n);  // mem,wp,bufo,bufw contiguous
  }
  // build fused, pre-transposed f16 weight matrix (L2-resident for the whole run)
  {
    int total = N_ * KTOT_;
    prep_B<<<(total + 255) / 256, 256, 0, stream>>>(w, w_in, w_signs, dmap, BmatT);
  }
  // 400 dependent fused steps
  for (int t = 0; t < T_; ++t) {
    step_k<<<N_ / 64, 256, 0, stream>>>(inputs, p, delays, BmatT,
                                        mem, wp, bufo, bufw, rec, t);
  }
  // readout
  readout1<<<(T_ * B_) / 4, 128, 0, stream>>>(rec, w_out, h2);
  readout2<<<(B_ * NOUT_ + 255) / 256, 256, 0, stream>>>(h2, out);
}
